// GraphSAGEEncoder_3066606649989
// MI455X (gfx1250) — compile-verified, matrix operand order checked by reading, not run on hardware
//
#include <hip/hip_runtime.h>
#include <hip/hip_bf16.h>

// ---------------------------------------------------------------------------
// GraphSAGE encoder for MI455X (gfx1250, wave32, WMMA).
//   h1 = relu(bn(mean_agg(x) @ W1_l + b1 + x @ W1_r))
//   out = l2norm_rows(mean_agg(h1) @ W2_l + b2 + h1 @ W2_r)
// Aggregation via fp32 global atomics (accumulators are L2-resident:
// 51MB + 102MB << 192MB L2). GEMMs via v_wmma_f32_16x16x32_f16 with
// concatenated A = [mean | x] against pre-transposed f16 weights.
// ---------------------------------------------------------------------------

typedef __attribute__((ext_vector_type(16))) _Float16 v16h;
typedef __attribute__((ext_vector_type(8)))  float    v8f;

union FragU { v16h v; unsigned u[8]; };

#define BN_EPS 1e-5f
#define L2_EPS 1e-12f

// ---- A fragment (16x32 f16, MxK) from an LDS panel sA[16][Kc] ------------
// ISA layout: lanes 0-15 hold row M=lane, K = {2v,2v+1} for VGPR v<4 and
// {16+2(v-4), ...} for v>=4; lanes 16-31 hold the same rows with K+8.
__device__ inline v16h load_a_frag_lds(const _Float16* sA, int Kc, int kk, int lane) {
  FragU f;
  const int m    = lane & 15;
  const int ksel = (lane >> 4) & 1;            // 0: K 0-7/16-23, 1: K 8-15/24-31
  const unsigned short* row = (const unsigned short*)(sA + m * Kc + kk * 32);
#pragma unroll
  for (int v = 0; v < 8; ++v) {
    const int kbase = ((v & 4) ? 16 : 0) + ksel * 8 + (v & 3) * 2;
    f.u[v] = *(const unsigned*)(row + kbase);  // halves (kbase, kbase+1), 4B aligned
  }
  return f.v;
}

// ---- B fragment (32x16 f16, KxN) from global WcT[n][k] (n-major) ----------
// Column-major B: lane 0-15 hold column N=lane (K 0-7/16-23), lanes 16-31
// hold the same columns with K+8. n-major storage makes every pair a dword.
__device__ inline v16h load_b_frag(const _Float16* __restrict__ WcT, int Kc, int kk,
                                   int ncolbase, int lane) {
  FragU f;
  const int n    = ncolbase + (lane & 15);
  const int ksel = (lane >> 4) & 1;
  const unsigned short* col = (const unsigned short*)(WcT + (size_t)n * Kc + kk * 32);
#pragma unroll
  for (int v = 0; v < 8; ++v) {
    const int kbase = ((v & 4) ? 16 : 0) + ksel * 8 + (v & 3) * 2;
    f.u[v] = *(const unsigned*)(col + kbase);
  }
  return f.v;
}

// ---------------------------------------------------------------------------
__global__ void zero_f4_kernel(float4* __restrict__ p, long long n4) {
  const float4 z = make_float4(0.f, 0.f, 0.f, 0.f);
  for (long long i = blockIdx.x * (long long)blockDim.x + threadIdx.x; i < n4;
       i += (long long)gridDim.x * blockDim.x)
    p[i] = z;
}

// Concatenate + transpose weights into f16 once per call.
// WcT1[n][k], k<128 -> W1_l[k][n], else W1_r[k-128][n]   (Kc=256)
// WcT2[n][k], k<256 -> W2_l[k][n], else W2_r[k-256][n]   (Kc=512)
__global__ void convert_weights_kernel(const float* __restrict__ W1l,
                                       const float* __restrict__ W1r,
                                       const float* __restrict__ W2l,
                                       const float* __restrict__ W2r,
                                       _Float16* __restrict__ WcT1,
                                       _Float16* __restrict__ WcT2) {
  const int tid = blockIdx.x * blockDim.x + threadIdx.x;
  if (tid < 256 * 256) {
    const int n = tid >> 8, k = tid & 255;
    const float v = (k < 128) ? W1l[k * 256 + n] : W1r[(k - 128) * 256 + n];
    WcT1[n * 256 + k] = (_Float16)v;
  }
  if (tid < 256 * 512) {
    const int n = tid >> 9, k = tid & 511;
    const float v = (k < 256) ? W2l[k * 256 + n] : W2r[(k - 256) * 256 + n];
    WcT2[n * 512 + k] = (_Float16)v;
  }
}

// Scatter-add: agg[dst] += feat[src] per feature, plus optional in-degree.
// F/4 lanes per edge, 4 float atomics per lane (atomics land in L2).
__global__ void scatter_add_kernel(const float* __restrict__ feat,
                                   const int* __restrict__ src,
                                   const int* __restrict__ dst,
                                   float* __restrict__ agg,
                                   float* __restrict__ cnt,
                                   int E, int F, int do_cnt) {
  const int gpe = F >> 2;                       // float4 groups per edge
  const long long total = (long long)E * gpe;
  for (long long t = blockIdx.x * (long long)blockDim.x + threadIdx.x; t < total;
       t += (long long)gridDim.x * blockDim.x) {
    const int e = (int)(t / gpe);
    const int g = (int)(t % gpe);
    const int s = src[e];
    const int d = dst[e];
    const float4 v = *(const float4*)(feat + (long long)s * F + g * 4);
    float* base = agg + (long long)d * F + g * 4;
    atomicAdd(base + 0, v.x);
    atomicAdd(base + 1, v.y);
    atomicAdd(base + 2, v.z);
    atomicAdd(base + 3, v.w);
    if (do_cnt && g == 0) atomicAdd(cnt + d, 1.0f);
  }
}

// ---- Fused GEMM 1: h = relu(bn([mean|x] @ [W1_l;W1_r] + b1)) --------------
// Block = 128 threads (4 waves); 16-row M-tile; wave w owns N-tiles 4w..4w+3.
__global__ __launch_bounds__(128) void gemm1_fused_kernel(
    const float* __restrict__ x, const float* __restrict__ agg1,
    const float* __restrict__ cnt, const _Float16* __restrict__ WcT1,
    const float* __restrict__ b1, const float* __restrict__ gamma,
    const float* __restrict__ beta, const float* __restrict__ bnmean,
    const float* __restrict__ bnvar, float* __restrict__ h, int Nn) {
  const int Kc = 256;
  __shared__ _Float16 sA[16 * 256];            // 8 KB A-panel [mean | x] in f16
  const int block_row = blockIdx.x * 16;
  const int tid = threadIdx.x;

  for (int idx = tid; idx < 16 * Kc; idx += 128) {
    const int m = idx >> 8;
    const int k = idx & 255;
    int row = block_row + m;
    if (row >= Nn) row = Nn - 1;
    float v;
    if (k < 128) {
      const float c = cnt[row];
      v = agg1[(long long)row * 128 + k] * (1.0f / fmaxf(c, 1.0f));
    } else {
      v = x[(long long)row * 128 + (k - 128)];
    }
    sA[m * Kc + k] = (_Float16)v;
  }
  __syncthreads();

  const int lane = tid & 31;
  const int wave = tid >> 5;
  v8f acc[4] = {};                             // 4 N-tiles of 16x16 f32 C/D

#pragma unroll
  for (int kk = 0; kk < 8; ++kk) {             // K = 256 in steps of 32
    const v16h a = load_a_frag_lds(sA, Kc, kk, lane);
#pragma unroll
    for (int t = 0; t < 4; ++t) {
      const v16h b = load_b_frag(WcT1, Kc, kk, (wave * 4 + t) * 16, lane);
      acc[t] = __builtin_amdgcn_wmma_f32_16x16x32_f16(
          false, a, false, b, (short)0, acc[t], false, false);
    }
  }

  // Epilogue: +bias, BN(eval), ReLU; C/D layout: VGPR r = rows r / r+8.
  const int rhalf = (lane >> 4) << 3;
#pragma unroll
  for (int t = 0; t < 4; ++t) {
    const int col = (wave * 4 + t) * 16 + (lane & 15);
    const float bia = b1[col];
    const float g = gamma[col], bt = beta[col], mu = bnmean[col];
    const float inv = rsqrtf(bnvar[col] + BN_EPS);
#pragma unroll
    for (int r = 0; r < 8; ++r) {
      const int row = block_row + r + rhalf;
      float v = acc[t][r] + bia;
      v = fmaf(g * (v - mu), inv, bt);
      v = fmaxf(v, 0.0f);
      if (row < Nn) h[(long long)row * 256 + col] = v;
    }
  }
}

// ---- Fused GEMM 2: out = l2norm_rows([mean2|h] @ [W2_l;W2_r] + b2) --------
// Block owns the full 16x256 output stripe -> row norms reduce in LDS.
__global__ __launch_bounds__(128) void gemm2_fused_kernel(
    const float* __restrict__ h, const float* __restrict__ agg2,
    const float* __restrict__ cnt, const _Float16* __restrict__ WcT2,
    const float* __restrict__ b2, float* __restrict__ out, int Nn) {
  const int Kc = 512;
  __shared__ _Float16 sA[16 * 512];            // 16 KB A-panel
  __shared__ float sNorm[16];
  const int block_row = blockIdx.x * 16;
  const int tid = threadIdx.x;

  if (tid < 16) sNorm[tid] = 0.0f;
  for (int idx = tid; idx < 16 * Kc; idx += 128) {
    const int m = idx >> 9;
    const int k = idx & 511;
    int row = block_row + m;
    if (row >= Nn) row = Nn - 1;
    float v;
    if (k < 256) {
      const float c = cnt[row];
      v = agg2[(long long)row * 256 + k] * (1.0f / fmaxf(c, 1.0f));
    } else {
      v = h[(long long)row * 256 + (k - 256)];
    }
    sA[m * Kc + k] = (_Float16)v;
  }
  __syncthreads();

  const int lane = tid & 31;
  const int wave = tid >> 5;
  v8f acc[4] = {};

#pragma unroll
  for (int kk = 0; kk < 16; ++kk) {            // K = 512 in steps of 32
    const v16h a = load_a_frag_lds(sA, Kc, kk, lane);
#pragma unroll
    for (int t = 0; t < 4; ++t) {
      const v16h b = load_b_frag(WcT2, Kc, kk, (wave * 4 + t) * 16, lane);
      acc[t] = __builtin_amdgcn_wmma_f32_16x16x32_f16(
          false, a, false, b, (short)0, acc[t], false, false);
    }
  }

  // Bias, accumulate row sum-of-squares in LDS, then normalize + store once.
  const int rhalf = (lane >> 4) << 3;
  float vals[4][8];
#pragma unroll
  for (int t = 0; t < 4; ++t) {
    const int col = (wave * 4 + t) * 16 + (lane & 15);
    const float bia = b2[col];
#pragma unroll
    for (int r = 0; r < 8; ++r) {
      const float v = acc[t][r] + bia;
      vals[t][r] = v;
      atomicAdd(&sNorm[r + rhalf], v * v);     // ds_add_f32
    }
  }
  __syncthreads();

#pragma unroll
  for (int t = 0; t < 4; ++t) {
    const int col = (wave * 4 + t) * 16 + (lane & 15);
#pragma unroll
    for (int r = 0; r < 8; ++r) {
      const int row = block_row + r + rhalf;
      const float scale = 1.0f / fmaxf(sqrtf(sNorm[r + rhalf]), L2_EPS);
      if (row < Nn) out[(long long)row * 256 + col] = vals[t][r] * scale;
    }
  }
}

// ---------------------------------------------------------------------------
extern "C" void kernel_launch(void* const* d_in, const int* in_sizes, int n_in,
                              void* d_out, int out_size, void* d_ws, size_t ws_size,
                              hipStream_t stream) {
  const float* x      = (const float*)d_in[0];
  const int*   edge   = (const int*)d_in[1];    // [2,E]: row0=src, row1=dst
  const float* W1l    = (const float*)d_in[2];
  const float* b1l    = (const float*)d_in[3];
  const float* W1r    = (const float*)d_in[4];
  const float* gamma  = (const float*)d_in[5];
  const float* beta   = (const float*)d_in[6];
  const float* bnmean = (const float*)d_in[7];
  const float* bnvar  = (const float*)d_in[8];
  const float* W2l    = (const float*)d_in[9];
  const float* b2l    = (const float*)d_in[10];
  const float* W2r    = (const float*)d_in[11];
  float* out = (float*)d_out;

  const int N = in_sizes[0] / 128;              // 100000
  const int E = in_sizes[1] / 2;                // 800000
  const int* src = edge;
  const int* dst = edge + E;

  // Workspace carve-up (all 16B-aligned: every region size % 16 floats == 0).
  float* agg1 = (float*)d_ws;                              // N*128 f32
  float* agg2 = agg1 + (size_t)N * 128;                    // N*256 f32
  float* cnt  = agg2 + (size_t)N * 256;                    // N     f32
  float* hbuf = cnt + N;                                   // N*256 f32
  _Float16* WcT1 = (_Float16*)(hbuf + (size_t)N * 256);    // 256*256 f16
  _Float16* WcT2 = WcT1 + 256 * 256;                       // 512*256 f16

  // 1) zero accumulators (agg1|agg2|cnt are contiguous)
  const long long zf = (long long)N * 128 + (long long)N * 256 + N;
  {
    const long long n4 = zf / 4;
    int blocks = (int)((n4 + 255) / 256);
    if (blocks > 65536) blocks = 65536;
    zero_f4_kernel<<<blocks, 256, 0, stream>>>((float4*)agg1, n4);
  }

  // 2) weights -> concatenated, transposed f16 (L2-resident thereafter)
  convert_weights_kernel<<<(256 * 512 + 255) / 256, 256, 0, stream>>>(
      W1l, W1r, W2l, W2r, WcT1, WcT2);

  // 3) layer-1 scatter-add + degree
  {
    const long long total = (long long)E * (128 / 4);
    int blocks = (int)((total + 255) / 256);
    if (blocks > 262144) blocks = 262144;
    scatter_add_kernel<<<blocks, 256, 0, stream>>>(x, src, dst, agg1, cnt, E, 128, 1);
  }

  // 4) fused GEMM-1 (WMMA) + bias + BN + ReLU -> hbuf
  gemm1_fused_kernel<<<(N + 15) / 16, 128, 0, stream>>>(
      x, agg1, cnt, WcT1, b1l, gamma, beta, bnmean, bnvar, hbuf, N);

  // 5) layer-2 scatter-add (same graph -> reuse cnt)
  {
    const long long total = (long long)E * (256 / 4);
    int blocks = (int)((total + 255) / 256);
    if (blocks > 262144) blocks = 262144;
    scatter_add_kernel<<<blocks, 256, 0, stream>>>(hbuf, src, dst, agg2, cnt, E, 256, 0);
  }

  // 6) fused GEMM-2 (WMMA) + bias + row L2-normalize -> out
  gemm2_fused_kernel<<<(N + 15) / 16, 128, 0, stream>>>(
      hbuf, agg2, cnt, WcT2, b2l, out, N);
}
